// PartitionFunction_83829171683881
// MI455X (gfx1250) — matrix operand order repaired and checked by reading, not compiled, not run on hardware
//
#include <hip/hip_runtime.h>
#include <cstddef>

typedef __attribute__((ext_vector_type(16))) _Float16 v16h;
typedef __attribute__((ext_vector_type(8)))  float    v8f;

// Problem constants (fixed by the reference: B=8, NT=T=128, S=256, N=41)
#define NB   8
#define NT   128
#define NNT  41
#define KTOT 49152          // 3 * NT*NT concatenated: [X_Y_Z | X_Y_z | X_y_Z]
#define NTILES_K 1536       // KTOT / 32
#define XT   8              // NT / 16
#define KC   6              // K chunks per row (8192 K each; 2 per quadrant)
#define TPC  256            // tiles per chunk
#define WV   8              // waves per matvec block
#define TPW  32             // tiles per wave
#define NEGV (-1000000000.0f)

// ---------------------------------------------------------------------------
// Deterministic 256-thread max reduction (identical code in both per-step
// kernels so every block derives bitwise-identical normalizers).
// ---------------------------------------------------------------------------
__device__ __forceinline__ float blockMax256(float v, float* buf, int tid) {
  buf[tid] = v; __syncthreads();
  for (int s = 128; s > 0; s >>= 1) {
    if (tid < s) buf[tid] = fmaxf(buf[tid], buf[tid + s]);
    __syncthreads();
  }
  float r = buf[0]; __syncthreads();
  return r;
}

// Loads rows 2..w-2 (nr = w-3 rows) raw into es[], s[w-1] into swv[], and
// computes m2 = max(es), msw = max(swv).
__device__ __forceinline__ void loadStats(const float* __restrict__ S, int b, int w,
                                          int nr, float (*es)[NT], float* swv,
                                          float* buf, int tid, float& m2, float& msw) {
  float lm = -INFINITY;
  for (int i = tid; i < nr * NT; i += 256) {
    float sv = S[((size_t)b * NNT + 2 + (i >> 7)) * NT + (i & 127)];
    es[i >> 7][i & 127] = sv;
    lm = fmaxf(lm, sv);
  }
  if (tid < NT) swv[tid] = S[((size_t)b * NNT + (w - 1)) * NT + tid];
  m2 = blockMax256(lm, buf, tid);
  float lw = (tid < NT) ? swv[tid] : -INFINITY;
  msw = blockMax256(lw, buf, tid);
}

// ---------------------------------------------------------------------------
// Preprocess: P[b][xt][kt][frag] = fp16(exp(rule quadrant)) in exact WMMA
// A-fragment order (16x32 tile = 512 halfs; lane L owns halfs L*16..L*16+15).
//   lane L<16 : row M=L,    halfs 0..7 -> K=0..7,  halfs 8..15 -> K=16..23
//   lane L>=16: row M=L-16, halfs 0..7 -> K=8..15, halfs 8..15 -> K=24..31
// ---------------------------------------------------------------------------
__global__ __launch_bounds__(256) void kPrep(const float* __restrict__ rule,
                                             _Float16* __restrict__ P) {
  int ktg = blockIdx.x, xt = blockIdx.y, b = blockIdx.z;
  size_t tileBase = ((size_t)(b * XT + xt) * NTILES_K + (size_t)ktg * 32);
  int tid = threadIdx.x;
  for (int i = tid; i < 32 * 512; i += 256) {
    int tl = i >> 9, off = i & 511;
    int L = off >> 4, h = off & 15;
    int row = L & 15;
    int hi = (L >> 4) ? 8 : 0;
    int kin = (h < 8) ? (hi + h) : (16 + hi + (h - 8));
    int kglob = (((ktg * 32 + tl) << 5) + kin);
    int q = kglob >> 14, rem = kglob & 16383;
    int r = rem >> 7, c = rem & 127;
    int R = (q == 2) ? (128 + r) : r;     // X_y_Z: terminal row
    int C = (q == 1) ? (128 + c) : c;     // X_Y_z: terminal col
    int x = xt * 16 + row;
    float val = rule[(((size_t)b * NT + x) * 256 + R) * 256 + C];
    P[(tileBase + tl) * 512 + off] = (_Float16)expf(val);
  }
}

// ---------------------------------------------------------------------------
// Init: s[b,2,x] = logsumexp over the T*T terminal-terminal block (== xyz)
// ---------------------------------------------------------------------------
__global__ __launch_bounds__(256) void kInit(const float* __restrict__ rule,
                                             float* __restrict__ S) {
  int x = blockIdx.x, b = blockIdx.y, tid = threadIdx.x;
  __shared__ float buf[256];
  const float* base = rule + (((size_t)b * NT + x) * 256 + 128) * 256 + 128;
  float acc = 0.f;
  for (int i = tid; i < 16384; i += 256)
    acc += expf(base[(size_t)(i >> 7) * 256 + (i & 127)]);   // entries <= 0
  buf[tid] = acc; __syncthreads();
  for (int s = 128; s > 0; s >>= 1) { if (tid < s) buf[tid] += buf[tid + s]; __syncthreads(); }
  if (tid == 0) S[((size_t)b * NNT + 2) * NT + x] = logf(buf[0]);
}

// ---------------------------------------------------------------------------
// Fused prepare + matvec for step w (computes partial dots for s[b,w,:]).
// Block (kc, xt, b), 256 threads = 8 waves.
//  Prologue: recompute normalizers; build this block's 8192-element exp-space
//    v-chunk directly in LDS:
//      kc 0..1 (YZ):  v = exp(2m2-mC) * sum_j exp(s[2+j,Y]-m2)*exp(s[w-2-j,Z]-m2)
//      kc 2..3 (Y,z): v = exp(s[w-1,Y]-mC)      (z-invariant)
//      kc 4..5 (y,Z): v = exp(s[w-1,Z]-mC)      (y-invariant)
//  Main: each wave streams 32 A-tiles (WMMA f32_16x16x32_f16, v in B col N=0;
//    B-layout: lane L holds B col N=L%16, 16 K values in table order; D col 0 in lanes 0/16).
//  The B fragment is double-buffered (bm0/bm1) so the per-iteration update of
//  all 16 elements never overlaps the B operand of the in-flight WMMA (avoids the
//  4-coexec WAR hazard NOPs from ISA 7.12.1).
// ---------------------------------------------------------------------------
__global__ __launch_bounds__(256) void kMatvec(const _Float16* __restrict__ P,
                                               const float* __restrict__ S,
                                               float* __restrict__ partial,
                                               int w) {
  __shared__ float es[37][NT];
  __shared__ float swv[NT];
  __shared__ float swe[NT];
  __shared__ float buf[256];
  __shared__ _Float16 vloc[8192];
  int kc = blockIdx.x, xt = blockIdx.y, b = blockIdx.z;
  int tid = threadIdx.x;
  int nr = w - 3;                 // rows 2..w-2 feed the YZ (split) term
  float m2, msw;
  loadStats(S, b, w, nr, es, swv, buf, tid, m2, msw);
  float mC = fmaxf(2.f * m2, msw);
  int q = kc >> 1;
  if (q == 0) {
    for (int i = tid; i < nr * NT; i += 256)
      es[i >> 7][i & 127] = expf(es[i >> 7][i & 127] - m2);
  } else {
    if (tid < NT) swe[tid] = expf(swv[tid] - mC);
  }
  __syncthreads();

  if (q == 0) {
    // register-tiled rank-nr outer product: thread owns a 4(Y) x 8(Z) tile
    float scale0 = expf(2.f * m2 - mC);
    int yg = tid >> 4, zg = tid & 15;
    int ly0 = yg * 4;                       // chunk-local Y base (0..63)
    int Yb = (kc & 1) * 64 + ly0;           // global Y base
    int Zb = zg * 8;
    float acc[4][8];
    #pragma unroll
    for (int u = 0; u < 4; ++u)
      #pragma unroll
      for (int v = 0; v < 8; ++v) acc[u][v] = 0.f;
    for (int j = 0; j < nr; ++j) {
      float ey[4], ez[8];
      #pragma unroll
      for (int u = 0; u < 4; ++u) ey[u] = es[j][Yb + u];
      #pragma unroll
      for (int v = 0; v < 8; ++v) ez[v] = es[nr - 1 - j][Zb + v];
      #pragma unroll
      for (int u = 0; u < 4; ++u)
        #pragma unroll
        for (int v = 0; v < 8; ++v) acc[u][v] += ey[u] * ez[v];
    }
    #pragma unroll
    for (int u = 0; u < 4; ++u)
      #pragma unroll
      for (int v = 0; v < 8; ++v)
        vloc[(ly0 + u) * 128 + Zb + v] = (_Float16)(acc[u][v] * scale0);
  } else if (q == 1) {
    int yoff = (kc & 1) << 6;
    for (int i = tid; i < 8192; i += 256) vloc[i] = (_Float16)swe[yoff + (i >> 7)];
  } else {
    for (int i = tid; i < 8192; i += 256) vloc[i] = (_Float16)swe[i & 127];
  }
  __syncthreads();

  int wv = tid >> 5, lane = tid & 31;
  const _Float16* Pb = P + ((size_t)(b * XT + xt) * NTILES_K + (size_t)kc * TPC) * 512;
#if defined(__gfx1250__)
  v8f acc8;
  #pragma unroll
  for (int i = 0; i < 8; ++i) acc8[i] = 0.f;
  v16h bm0, bm1;                             // double-buffered B fragments
  #pragma unroll
  for (int i = 0; i < 16; ++i) { bm0[i] = (_Float16)0.f; bm1[i] = (_Float16)0.f; }
  const _Float16* apBase = Pb + (size_t)wv * TPW * 512 + lane * 16;
  const _Float16* vBase  = vloc + (size_t)wv * TPW * 32 + (lane >> 4) * 8;
  for (int t = 0; t < TPW; t += 2) {
    const _Float16* ap0 = apBase + (size_t)t * 512;
    v16h a0 = *(const v16h*)ap0;             // 32B coalesced fragment (2x b128)
    __builtin_prefetch(ap0 + 8 * 512, 0, 1); // speculative, unconditional
    for (int s = 0; s < 16; ++s) bm0[s] = vBase[t * 32 + (s < 8 ? s : s + 8)];  // B column = v (from LDS), table k order
    acc8 = __builtin_amdgcn_wmma_f32_16x16x32_f16(
        false, a0, false, bm0, (short)0, acc8, false, false);

    const _Float16* ap1 = ap0 + 512;
    v16h a1 = *(const v16h*)ap1;
    __builtin_prefetch(ap1 + 8 * 512, 0, 1);
    for (int s = 0; s < 16; ++s) bm1[s] = vBase[(t + 1) * 32 + (s < 8 ? s : s + 8)];
    acc8 = __builtin_amdgcn_wmma_f32_16x16x32_f16(
        false, a1, false, bm1, (short)0, acc8, false, false);
  }
  if ((lane & 15) == 0) {
    int mb = (lane == 0) ? 0 : 8;            // lane0: M=0..7, lane16: M=8..15
    float* po = partial + (((size_t)(b * XT + xt) * KC + kc) * WV + wv) * 16 + mb;
    #pragma unroll
    for (int r = 0; r < 8; ++r) po[r] = acc8[r];
  }
#else
  // host-pass fallback (never executed on device)
  if (lane == 0 && wv == 0) {
    for (int m = 0; m < 16; ++m) {
      float s = 0.f;
      for (int k = 0; k < 8192; ++k) s += (float)vloc[k];
      partial[(((size_t)(b * XT + xt) * KC + kc) * WV) * 16 + m] = s;
    }
  }
#endif
}

// ---------------------------------------------------------------------------
// Reduce: s[b,w,x] = mC + log( sum over 48 partials ).  Recomputes mC with
// the identical deterministic reduction used in kMatvec.
// ---------------------------------------------------------------------------
__global__ __launch_bounds__(256) void kReduce(float* __restrict__ S,
                                               const float* __restrict__ partial,
                                               int w) {
  __shared__ float es[37][NT];
  __shared__ float swv[NT];
  __shared__ float buf[256];
  int b = blockIdx.x, tid = threadIdx.x;
  int nr = w - 3;
  float m2, msw;
  loadStats(S, b, w, nr, es, swv, buf, tid, m2, msw);
  float mC = fmaxf(2.f * m2, msw);
  if (tid < NT) {
    float sum = 0.f;
    const float* pin = partial + ((size_t)(b * XT + (tid >> 4)) * KC * WV) * 16 + (tid & 15);
    #pragma unroll 4
    for (int u = 0; u < KC * WV; ++u) sum += pin[u * 16];
    float xw = (sum > 0.f) ? (mC + logf(sum)) : NEGV;
    if (!(xw > NEGV)) xw = NEGV;
    S[((size_t)b * NNT + w) * NT + tid] = xw;
  }
}

// ---------------------------------------------------------------------------
// Final: out[b] = logsumexp_x( s[b, lens[b], x] + root[b, x] )
// ---------------------------------------------------------------------------
__global__ __launch_bounds__(128) void kFinal(const float* __restrict__ S,
                                              const float* __restrict__ root,
                                              const int* __restrict__ lens,
                                              float* __restrict__ out) {
  int b = blockIdx.x, tid = threadIdx.x;
  __shared__ float buf[128];
  int len = lens[b];
  float v = S[((size_t)b * NNT + len) * NT + tid] + root[(size_t)b * NT + tid];
  buf[tid] = v; __syncthreads();
  for (int s = 64; s > 0; s >>= 1) { if (tid < s) buf[tid] = fmaxf(buf[tid], buf[tid + s]); __syncthreads(); }
  float m = buf[0]; __syncthreads();
  buf[tid] = expf(v - m); __syncthreads();
  for (int s = 64; s > 0; s >>= 1) { if (tid < s) buf[tid] += buf[tid + s]; __syncthreads(); }
  if (tid == 0) out[b] = m + logf(buf[0]);
}

// ---------------------------------------------------------------------------
// Host driver.  Workspace layout (bytes):
//   P       : 64 * 1536 * 512 halfs = 100,663,296   (fits MI455X 192MB L2)
//   S       : 8 * 41 * 128 floats   =     167,936
//   partial : 64 * 6 * 8 * 16 f32   =     196,608
// 79 launches total, all stream-ordered and graph-capturable.
// ---------------------------------------------------------------------------
extern "C" void kernel_launch(void* const* d_in, const int* in_sizes, int n_in,
                              void* d_out, int out_size, void* d_ws, size_t ws_size,
                              hipStream_t stream) {
  (void)in_sizes; (void)n_in; (void)out_size; (void)ws_size;
  const float* rule = (const float*)d_in[0];
  const float* root = (const float*)d_in[1];
  const int*   lens = (const int*)d_in[2];
  float* out = (float*)d_out;

  char* ws = (char*)d_ws;
  _Float16* P = (_Float16*)(ws);
  float*    S = (float*)(ws + (size_t)100663296);
  float*    Q = (float*)(ws + (size_t)100663296 + 167936);

  kPrep<<<dim3(48, XT, NB), 256, 0, stream>>>(rule, P);
  kInit<<<dim3(NT, NB), 256, 0, stream>>>(rule, S);
  for (int w = 3; w <= 40; ++w) {
    kMatvec<<<dim3(KC, XT, NB), 256, 0, stream>>>(P, S, Q, w);
    kReduce<<<NB, 256, 0, stream>>>(S, Q, w);
  }
  kFinal<<<NB, 128, 0, stream>>>(S, root, lens, out);
}
